// L2Attention_7808250544321
// MI455X (gfx1250) — matrix operand order repaired and checked by reading, not compiled, not run on hardware
//
#include <hip/hip_runtime.h>

// ---------------------------------------------------------------------------
// L2 attention for MI455X (gfx1250, wave32, WMMA bf16 16x16x32, f32 accum)
// Data movement: GLOBAL_LOAD_ASYNC_TO_LDS_B128 (ASYNCcnt) double-buffering.
// ---------------------------------------------------------------------------

typedef __bf16 bf16_t;
typedef __attribute__((ext_vector_type(16))) __bf16 v16bf;
typedef __attribute__((ext_vector_type(8)))  __bf16 v8bf;
typedef __attribute__((ext_vector_type(8)))  float  v8f;

#define DIMD   1024
#define SEQN   2048
#define HEADS  16
#define DHEAD  64
#define BATCH  2
#define NTOK   (BATCH * SEQN)          // 4096 rows
#define SCALE  0.125f                  // dh^-0.5

// one 16B async global->LDS copy; OFF applies to both global and LDS address
#define ASYNC_LDS16(lds32, ga64, OFF)                                        \
  asm volatile("global_load_async_to_lds_b128 %0, %1, off offset:" #OFF     \
               :: "v"(lds32), "v"(ga64) : "memory")

#define WAIT_ASYNC(N)                                                        \
  asm volatile("s_wait_asynccnt " #N ::: "memory")

__device__ __forceinline__ v16bf mk_frag(v8bf lo, v8bf hi) {
  v16bf r;
#pragma unroll
  for (int i = 0; i < 8; ++i) { r[i] = lo[i]; r[i + 8] = hi[i]; }
  return r;
}
__device__ __forceinline__ v8bf lds_v8(const bf16_t* p) {
  return *(const v8bf*)p;
}

// ---------------------------------------------------------------------------
// fp32 -> bf16 conversion
// ---------------------------------------------------------------------------
__global__ __launch_bounds__(256) void cvt_bf16_kernel(const float* __restrict__ in,
                                                       bf16_t* __restrict__ out, int n) {
  int i = blockIdx.x * 256 + threadIdx.x;
  if (i < n) out[i] = (bf16_t)in[i];
}

// ---------------------------------------------------------------------------
// Row sum-of-squares over dh=64 for every (b,h,n) row of q
// ---------------------------------------------------------------------------
__global__ __launch_bounds__(256) void sq_kernel(const bf16_t* __restrict__ q,
                                                 float* __restrict__ sq, int rows) {
  int r = blockIdx.x * 256 + threadIdx.x;
  if (r >= rows) return;
  const bf16_t* p = q + (size_t)r * DHEAD;
  float s = 0.f;
#pragma unroll
  for (int i = 0; i < DHEAD; ++i) { float t = (float)p[i]; s += t * t; }
  sq[r] = s;
}

// ---------------------------------------------------------------------------
// bf16 GEMM:  C[M,N] = A[M,K] * B[N,K]^T   (row-major, K inner)
// Block tile 128x128, 8 waves (2x4), wave tile 64x32, K-step 64,
// double-buffered LDS filled by async global->LDS copies.
// EPI==0: scatter columns into q/v buffers laid out (b,h,n,dh), bf16.
// EPI==1: plain fp32 row-major store.
// ---------------------------------------------------------------------------
template <int EPI>
__global__ __launch_bounds__(256) void gemm_bf16_kernel(
    const bf16_t* __restrict__ A, const bf16_t* __restrict__ Bw,
    bf16_t* __restrict__ out_q, bf16_t* __restrict__ out_v,
    float* __restrict__ out_f, int K, int Nn) {
  const int LW = 72;                       // 64 + 8 pad (144B rows, 16B aligned)
  __shared__ bf16_t As[2][128 * LW];
  __shared__ bf16_t Bs[2][128 * LW];

  const int tid  = threadIdx.x;
  const int bm   = blockIdx.x;
  const int bn   = blockIdx.y;
  const int w    = tid >> 5;
  const int lane = tid & 31;
  const int l16  = lane & 15;
  const int hi   = lane >> 4;
  const int wm   = w >> 2;                 // 0..1 -> 64-row strip
  const int wn   = w & 3;                  // 0..3 -> 32-col strip

  // staging assignment: 2 threads per row, 32 bf16 (64B = 4x16B) each
  const int srow = tid >> 1;
  const int scb  = (tid & 1) * 32;
  const bf16_t* gA = A  + (size_t)(bm * 128 + srow) * K + scb;
  const bf16_t* gB = Bw + (size_t)(bn * 128 + srow) * K + scb;
  const uint32_t laA0 = (uint32_t)(uintptr_t)(&As[0][srow * LW + scb]);
  const uint32_t laB0 = (uint32_t)(uintptr_t)(&Bs[0][srow * LW + scb]);
  const uint32_t laA1 = (uint32_t)(uintptr_t)(&As[1][srow * LW + scb]);
  const uint32_t laB1 = (uint32_t)(uintptr_t)(&Bs[1][srow * LW + scb]);

  auto issue_stage = [&](int buf, int k0) {
    uint64_t ga = (uint64_t)(uintptr_t)(gA + k0);
    uint64_t gb = (uint64_t)(uintptr_t)(gB + k0);
    uint32_t la = buf ? laA1 : laA0;
    uint32_t lb = buf ? laB1 : laB0;
    ASYNC_LDS16(la, ga, 0);  ASYNC_LDS16(la, ga, 16);
    ASYNC_LDS16(la, ga, 32); ASYNC_LDS16(la, ga, 48);
    ASYNC_LDS16(lb, gb, 0);  ASYNC_LDS16(lb, gb, 16);
    ASYNC_LDS16(lb, gb, 32); ASYNC_LDS16(lb, gb, 48);
  };

  v8f acc[4][2];
#pragma unroll
  for (int mt = 0; mt < 4; ++mt)
#pragma unroll
    for (int nt = 0; nt < 2; ++nt) acc[mt][nt] = 0.0f;

  const int NK = K / 64;
  issue_stage(0, 0);

  for (int k = 0; k < NK; ++k) {
    const int cur = k & 1;
    if (k + 1 < NK) {
      issue_stage(cur ^ 1, (k + 1) * 64);
      WAIT_ASYNC(0x8);                     // stage k landed; k+1 in flight
    } else {
      WAIT_ASYNC(0x0);
    }
    __syncthreads();

    const bf16_t* Ab = As[cur];
    const bf16_t* Bb = Bs[cur];
#pragma unroll
    for (int ks = 0; ks < 2; ++ks) {
      v16bf af[4];
#pragma unroll
      for (int mt = 0; mt < 4; ++mt) {
        const bf16_t* p = Ab + (wm * 64 + mt * 16 + l16) * LW + ks * 32;
        af[mt] = mk_frag(lds_v8(p + hi * 8), lds_v8(p + 16 + hi * 8));
      }
      v16bf bfr[2];
#pragma unroll
      for (int nt = 0; nt < 2; ++nt) {
        const bf16_t* p = Bb + (wn * 32 + nt * 16 + l16) * LW + ks * 32;
        bfr[nt] = mk_frag(lds_v8(p + hi * 8), lds_v8(p + 16 + hi * 8));
      }
#pragma unroll
      for (int mt = 0; mt < 4; ++mt)
#pragma unroll
        for (int nt = 0; nt < 2; ++nt)
          acc[mt][nt] = __builtin_amdgcn_wmma_f32_16x16x32_bf16(
              false, af[mt], false, bfr[nt], (short)0, acc[mt][nt], false, false);
    }
    __syncthreads();                       // before next issue overwrites buf
  }

  // ---- epilogue: C-layout, lane holds col l16, rows i + hi*8 ----
#pragma unroll
  for (int mt = 0; mt < 4; ++mt)
#pragma unroll
    for (int nt = 0; nt < 2; ++nt) {
      int col     = bn * 128 + wn * 32 + nt * 16 + l16;
      int rowBase = bm * 128 + wm * 64 + mt * 16 + hi * 8;
#pragma unroll
      for (int i = 0; i < 8; ++i) {
        int r = rowBase + i;
        float val = acc[mt][nt][i];
        if (EPI == 0) {
          int bb = r >> 11, n = r & (SEQN - 1);
          bf16_t hv = (bf16_t)val;
          if (col < DIMD) {
            int h = col >> 6, dh = col & 63;
            out_q[(((size_t)(bb * HEADS + h)) * SEQN + n) * DHEAD + dh] = hv;
          } else {
            int e = col - DIMD;
            int h = e >> 6, dh = e & 63;
            out_v[(((size_t)(bb * HEADS + h)) * SEQN + n) * DHEAD + dh] = hv;
          }
        } else {
          out_f[(size_t)r * Nn + col] = val;
        }
      }
    }
}

// ---------------------------------------------------------------------------
// Flash-style L2 attention.  Grid: (n/128 query blocks, b*h).  256 threads.
// Each wave owns 16 query rows x all 128 keys of the current key block.
// scores = (2*QQ^T - |q_i|^2 - |q_j|^2) * scale, online softmax, O += P V.
// K tile staged with async global->LDS, overlapped with V transpose.
// ---------------------------------------------------------------------------
__global__ __launch_bounds__(256) void l2attn_kernel(
    const bf16_t* __restrict__ q, const bf16_t* __restrict__ v,
    const float* __restrict__ sqg, bf16_t* __restrict__ obuf) {
  const int QP = 72;                       // 64 + 8 pad
  const int VP = 136;                      // 128 + 8 pad
  __shared__ bf16_t Qs[128 * QP];
  __shared__ bf16_t Ks[128 * QP];
  __shared__ bf16_t VTs[DHEAD * VP];       // transposed V tile [dh][key]
  __shared__ bf16_t Ps[8 * 16 * VP];       // wave-private P staging

  const int tid  = threadIdx.x;
  const int w    = tid >> 5;
  const int lane = tid & 31;
  const int l16  = lane & 15;
  const int hi   = lane >> 4;
  const int qb   = blockIdx.x;             // query block (0..15)
  const int bh   = blockIdx.y;             // 0..31

  const bf16_t* qbase = q + (size_t)bh * SEQN * DHEAD;
  const bf16_t* vbase = v + (size_t)bh * SEQN * DHEAD;
  const float*  sqb   = sqg + (size_t)bh * SEQN;

  const int srow = tid >> 1;
  const int scb  = (tid & 1) * 32;

  {   // async load 128x64 query block (each thread 32 bf16 = 4x16B)
    uint64_t ga = (uint64_t)(uintptr_t)(qbase + (size_t)(qb * 128 + srow) * DHEAD + scb);
    uint32_t la = (uint32_t)(uintptr_t)(Qs + srow * QP + scb);
    ASYNC_LDS16(la, ga, 0);  ASYNC_LDS16(la, ga, 16);
    ASYNC_LDS16(la, ga, 32); ASYNC_LDS16(la, ga, 48);
  }

  float sqq[8];
  {
    int rb = qb * 128 + w * 16 + hi * 8;
#pragma unroll
    for (int i = 0; i < 8; ++i) sqq[i] = sqb[rb + i];
  }

  float mrow[8], lrow[8];
  v8f oacc[4];
#pragma unroll
  for (int i = 0; i < 8; ++i) { mrow[i] = -1e30f; lrow[i] = 0.f; }
#pragma unroll
  for (int dt = 0; dt < 4; ++dt) oacc[dt] = 0.0f;

  for (int j = 0; j < SEQN / 128; ++j) {
    __syncthreads();   // previous iteration done with Ks/VTs (Qs in flight, j==0)
    {   // async stage key block (keys are q rows)
      uint64_t ga = (uint64_t)(uintptr_t)(qbase + (size_t)(j * 128 + srow) * DHEAD + scb);
      uint32_t la = (uint32_t)(uintptr_t)(Ks + srow * QP + scb);
      ASYNC_LDS16(la, ga, 0);  ASYNC_LDS16(la, ga, 16);
      ASYNC_LDS16(la, ga, 32); ASYNC_LDS16(la, ga, 48);
    }
    {   // V block, transposed into LDS [dh][key] (overlaps async K tile)
      int key = tid >> 1;
      int db  = (tid & 1) * 32;
      const bf16_t* src = vbase + (size_t)(j * 128 + key) * DHEAD + db;
#pragma unroll
      for (int t = 0; t < 32; ++t) VTs[(db + t) * VP + key] = src[t];
    }
    if (j + 1 < SEQN / 128) {              // prefetch next V block into L2/L0
      __builtin_prefetch((const char*)(vbase + (size_t)((j + 1) * 128) * DHEAD) +
                             (size_t)tid * 64, 0, 1);
    }
    WAIT_ASYNC(0x0);
    __syncthreads();

    float sqk[8];
#pragma unroll
    for (int nt = 0; nt < 8; ++nt) sqk[nt] = sqb[j * 128 + nt * 16 + l16];

    // ---- S = Q Kt : wave rows w*16..w*16+15, 8 N-tiles over 128 keys ----
    v16bf af[2];
#pragma unroll
    for (int ks = 0; ks < 2; ++ks) {
      const bf16_t* p = Qs + (w * 16 + l16) * QP + ks * 32;
      af[ks] = mk_frag(lds_v8(p + hi * 8), lds_v8(p + 16 + hi * 8));
    }
    float s[8][8];
#pragma unroll
    for (int nt = 0; nt < 8; ++nt) {
      v8f c = 0.0f;
#pragma unroll
      for (int ks = 0; ks < 2; ++ks) {
        const bf16_t* pb = Ks + (nt * 16 + l16) * QP + ks * 32;
        v16bf bfr = mk_frag(lds_v8(pb + hi * 8), lds_v8(pb + 16 + hi * 8));
        c = __builtin_amdgcn_wmma_f32_16x16x32_bf16(
            false, af[ks], false, bfr, (short)0, c, false, false);
      }
#pragma unroll
      for (int i = 0; i < 8; ++i)
        s[nt][i] = (2.0f * c[i] - sqq[i] - sqk[nt]) * SCALE;
    }

    // ---- online softmax (row stats via 16-lane shfl_xor reduction) ----
    float alpha[8];
#pragma unroll
    for (int i = 0; i < 8; ++i) {
      float mx = s[0][i];
#pragma unroll
      for (int nt = 1; nt < 8; ++nt) mx = fmaxf(mx, s[nt][i]);
      mx = fmaxf(mx, __shfl_xor(mx, 1));
      mx = fmaxf(mx, __shfl_xor(mx, 2));
      mx = fmaxf(mx, __shfl_xor(mx, 4));
      mx = fmaxf(mx, __shfl_xor(mx, 8));
      float mn = fmaxf(mrow[i], mx);
      alpha[i] = __expf(mrow[i] - mn);
      mrow[i]  = mn;
    }
#pragma unroll
    for (int nt = 0; nt < 8; ++nt)
#pragma unroll
      for (int i = 0; i < 8; ++i) s[nt][i] = __expf(s[nt][i] - mrow[i]);
#pragma unroll
    for (int i = 0; i < 8; ++i) {
      float t = 0.f;
#pragma unroll
      for (int nt = 0; nt < 8; ++nt) t += s[nt][i];
      t += __shfl_xor(t, 1);
      t += __shfl_xor(t, 2);
      t += __shfl_xor(t, 4);
      t += __shfl_xor(t, 8);
      lrow[i] = lrow[i] * alpha[i] + t;
    }
#pragma unroll
    for (int dt = 0; dt < 4; ++dt)
#pragma unroll
      for (int i = 0; i < 8; ++i) oacc[dt][i] *= alpha[i];

    // ---- P (C-layout) -> wave-private LDS (A-layout transpose) ----
    bf16_t* pl = Ps + w * 16 * VP;
#pragma unroll
    for (int nt = 0; nt < 8; ++nt)
#pragma unroll
      for (int i = 0; i < 8; ++i)
        pl[(hi * 8 + i) * VP + nt * 16 + l16] = (bf16_t)s[nt][i];

    // ---- O += P V  (K = 128 keys, 4 k-steps; N = dh, 4 tiles) ----
#pragma unroll
    for (int ks = 0; ks < 4; ++ks) {
      const bf16_t* pa = pl + l16 * VP + ks * 32;
      v16bf paf = mk_frag(lds_v8(pa + hi * 8), lds_v8(pa + 16 + hi * 8));
#pragma unroll
      for (int dt = 0; dt < 4; ++dt) {
        const bf16_t* pb = VTs + (dt * 16 + l16) * VP + ks * 32 + hi * 8;
        v16bf pbf = mk_frag(lds_v8(pb), lds_v8(pb + 16));
        oacc[dt] = __builtin_amdgcn_wmma_f32_16x16x32_bf16(
            false, paf, false, pbf, (short)0, oacc[dt], false, false);
      }
    }
  }

  // ---- normalize and store to (b, n, h*64+dh) bf16 buffer ----
  int b = bh >> 4, h = bh & 15;
#pragma unroll
  for (int dt = 0; dt < 4; ++dt)
#pragma unroll
    for (int i = 0; i < 8; ++i) {
      float o   = oacc[dt][i] / lrow[i];
      int rowN  = qb * 128 + w * 16 + hi * 8 + i;
      int col   = h * DHEAD + dt * 16 + l16;
      obuf[(size_t)(b * SEQN + rowN) * DIMD + col] = (bf16_t)o;
    }
}

// ---------------------------------------------------------------------------
extern "C" void kernel_launch(void* const* d_in, const int* in_sizes, int n_in,
                              void* d_out, int out_size, void* d_ws, size_t ws_size,
                              hipStream_t stream) {
  const float* x   = (const float*)d_in[0];   // (2,2048,1024)
  const float* Wqv = (const float*)d_in[1];   // (2048,1024)
  const float* Wo  = (const float*)d_in[2];   // (1024,1024)
  float* out = (float*)d_out;                 // (2,2048,1024)

  char* ws = (char*)d_ws;
  size_t off = 0;
  bf16_t* xb   = (bf16_t*)(ws + off); off += (size_t)NTOK * DIMD * 2;        // 8 MB
  bf16_t* wqvb = (bf16_t*)(ws + off); off += (size_t)2 * DIMD * DIMD * 2;    // 4 MB
  bf16_t* wob  = (bf16_t*)(ws + off); off += (size_t)DIMD * DIMD * 2;        // 2 MB
  bf16_t* qbuf = (bf16_t*)(ws + off); off += (size_t)NTOK * DIMD * 2;        // 8 MB
  bf16_t* vbuf = (bf16_t*)(ws + off); off += (size_t)NTOK * DIMD * 2;        // 8 MB
  float*  sqg  = (float*)(ws + off);  off += (size_t)BATCH * HEADS * SEQN * 4;
  bf16_t* obuf = (bf16_t*)(ws + off); off += (size_t)NTOK * DIMD * 2;        // 8 MB

  const int nx  = NTOK * DIMD;         // 4,194,304
  const int nqv = 2 * DIMD * DIMD;     // 2,097,152
  const int nwo = DIMD * DIMD;         // 1,048,576
  cvt_bf16_kernel<<<(nx  + 255) / 256, 256, 0, stream>>>(x,   xb,   nx);
  cvt_bf16_kernel<<<(nqv + 255) / 256, 256, 0, stream>>>(Wqv, wqvb, nqv);
  cvt_bf16_kernel<<<(nwo + 255) / 256, 256, 0, stream>>>(Wo,  wob,  nwo);

  // qv = x @ Wqv^T, scattered into q/v (b,h,n,dh) bf16
  gemm_bf16_kernel<0><<<dim3(NTOK / 128, (2 * DIMD) / 128), 256, 0, stream>>>(
      xb, wqvb, qbuf, vbuf, nullptr, DIMD, 2 * DIMD);

  sq_kernel<<<(BATCH * HEADS * SEQN) / 256, 256, 0, stream>>>(
      qbuf, sqg, BATCH * HEADS * SEQN);

  l2attn_kernel<<<dim3(SEQN / 128, BATCH * HEADS), 256, 0, stream>>>(
      qbuf, vbuf, sqg, obuf);

  // final = attn_out @ Wo^T, fp32 out
  gemm_bf16_kernel<1><<<dim3(NTOK / 128, DIMD / 128), 256, 0, stream>>>(
      obuf, wob, nullptr, nullptr, out, DIMD, DIMD);
}
